// MultiHeadSelfAttention_86242943303849
// MI455X (gfx1250) — compile-verified, matrix operand order checked by reading, not run on hardware
//
#include <hip/hip_runtime.h>
#include <hip/hip_bf16.h>

#define SEQ 4096
#define DMODEL 1024
#define NHEAD 16
#define HDIM 64

typedef __attribute__((ext_vector_type(16))) _Float16 v16h;
typedef __attribute__((ext_vector_type(8)))  _Float16 v8h;
typedef __attribute__((ext_vector_type(8)))  float    v8f;

// D = A(16x32 f16) * B(32x16 f16) + C(16x16 f32)
__device__ __forceinline__ v8f wmma_f16(v16h a, v16h b, v8f c) {
  return __builtin_amdgcn_wmma_f32_16x16x32_f16(
      /*neg_a=*/false, a, /*neg_b=*/false, b,
      /*c_mod=*/(short)0, c, /*reuse_a=*/false, /*reuse_b=*/false);
}

// Load one f16 A/B fragment per the CDNA5 16-bit layout.
// Caller passes p = row_base + ko (ko = 8 for lanes 16..31, else 0);
// fragment = halves [p .. p+7] and [p+16 .. p+23].
__device__ __forceinline__ v16h load_frag16(const _Float16* p) {
  v8h lo = *(const v8h*)(p);
  v8h hi = *(const v8h*)(p + 16);
  return __builtin_shufflevector(lo, hi, 0,1,2,3,4,5,6,7,8,9,10,11,12,13,14,15);
}

// Asynchronously copy one contiguous 4KB block (global f16 -> LDS) using
// GLOBAL_LOAD_ASYNC_TO_LDS_B128 (ASYNCcnt-tracked DMA into LDS, CDNA5).
// 32 lanes x 8 x 16B. Generic LDS pointer low 32 bits == LDS byte offset.
__device__ __forceinline__ void async_copy_4k(const _Float16* gsrc,
                                              _Float16* ldst, int lane) {
  unsigned lds_addr = (unsigned)(uintptr_t)ldst + (unsigned)lane * 16u;
  unsigned long long gaddr =
      (unsigned long long)(uintptr_t)gsrc + (unsigned long long)lane * 16ull;
#pragma unroll
  for (int i = 0; i < 8; ++i) {
    asm volatile("global_load_async_to_lds_b128 %0, %1, off"
                 :: "v"(lds_addr + (unsigned)i * 512u),
                    "v"(gaddr + (unsigned long long)i * 512ull)
                 : "memory");
  }
}

// ---------------------------------------------------------------------------
// C[4096,1024] = A[4096,1024] (row-major fp32) * W[1024,1024]^T (fp32).
// Block = 256 threads = 8 waves. Block tile 128(M) x 64(N), K-step 32.
// Each wave computes a 16x64 strip (4 accumulator tiles).
// ---------------------------------------------------------------------------
__global__ __launch_bounds__(256) void gemm_xwT(const float* __restrict__ A,
                                                const float* __restrict__ W,
                                                float* __restrict__ C) {
  __shared__ __align__(16) _Float16 ldsA[128 * 32];
  __shared__ __align__(16) _Float16 ldsB[64 * 32];

  const int tid  = threadIdx.x;
  const int m0   = blockIdx.x * 128;
  const int n0   = blockIdx.y * 64;
  const int lane = tid & 31;
  const int wv   = tid >> 5;
  const int hi   = lane >> 4;    // lane half
  const int ko   = hi * 8;       // K-pattern offset
  const int ln   = lane & 15;

  v8f acc[4] = {};

  const int arow = tid >> 1;           // 0..127
  const int acol = (tid & 1) * 16;     // 0 or 16
  const int brow = tid >> 2;           // 0..63
  const int bcol = (tid & 3) * 8;      // 0,8,16,24

  for (int k0 = 0; k0 < DMODEL; k0 += 32) {
    // Stage A tile (128x32) as f16
    {
      const float* src = A + (size_t)(m0 + arow) * DMODEL + k0 + acol;
      __builtin_prefetch(src + 32, 0, 0);   // next k-step
      __align__(16) _Float16 tmp[16];
#pragma unroll
      for (int i = 0; i < 16; i += 4) {
        float4 f = *(const float4*)(src + i);
        tmp[i+0] = (_Float16)f.x; tmp[i+1] = (_Float16)f.y;
        tmp[i+2] = (_Float16)f.z; tmp[i+3] = (_Float16)f.w;
      }
      _Float16* dst = &ldsA[arow * 32 + acol];
      *(v8h*)(dst)     = *(const v8h*)(tmp);
      *(v8h*)(dst + 8) = *(const v8h*)(tmp + 8);
    }
    // Stage B tile: ldsB[n][k] = W[n0+n][k0+k]  (B^T, rows contiguous in K)
    {
      const float* src = W + (size_t)(n0 + brow) * DMODEL + k0 + bcol;
      __builtin_prefetch(src + 32, 0, 0);
      __align__(16) _Float16 tmp[8];
#pragma unroll
      for (int i = 0; i < 8; i += 4) {
        float4 f = *(const float4*)(src + i);
        tmp[i+0] = (_Float16)f.x; tmp[i+1] = (_Float16)f.y;
        tmp[i+2] = (_Float16)f.z; tmp[i+3] = (_Float16)f.w;
      }
      *(v8h*)(&ldsB[brow * 32 + bcol]) = *(const v8h*)(tmp);
    }
    __syncthreads();

    v16h a = load_frag16(&ldsA[(wv * 16 + ln) * 32 + ko]);
    v16h b[4];
#pragma unroll
    for (int j = 0; j < 4; ++j)
      b[j] = load_frag16(&ldsB[(j * 16 + ln) * 32 + ko]);
#pragma unroll
    for (int j = 0; j < 4; ++j)
      acc[j] = wmma_f16(a, b[j], acc[j]);
    __syncthreads();
  }

  // Epilogue: C/D layout — lane = col, VGPR r = row (r + 8*hi)
#pragma unroll
  for (int j = 0; j < 4; ++j)
#pragma unroll
    for (int r = 0; r < 8; ++r) {
      int m = m0 + wv * 16 + r + hi * 8;
      int n = n0 + j * 16 + ln;
      C[(size_t)m * DMODEL + n] = acc[j][r];
    }
}

// ---------------------------------------------------------------------------
// RoPE(Q,K) -> f16 [H][S][HDIM];  V -> f16 transposed [H][HDIM][S]
// ---------------------------------------------------------------------------
__global__ __launch_bounds__(256) void rope_convert(
    const float* __restrict__ Q32, const float* __restrict__ K32,
    const float* __restrict__ V32, const int* __restrict__ pos,
    _Float16* __restrict__ Q16, _Float16* __restrict__ K16,
    _Float16* __restrict__ Vt16) {
  int idx = blockIdx.x * 256 + threadIdx.x;   // 0 .. SEQ*DMODEL-1
  int s  = idx >> 10;
  int d  = idx & (DMODEL - 1);
  int h  = d >> 6;
  int dh = d & (HDIM - 1);

  // V: convert + transpose so PV GEMM B-rows are contiguous over keys
  Vt16[((size_t)(h * HDIM + dh)) * SEQ + s] = (_Float16)V32[idx];

  if ((dh & 1) == 0) {
    int j = dh >> 1;                       // pair index, 0..31
    float inv = __powf(10000.0f, -(float)j * (1.0f / 32.0f));
    float ang = (float)pos[s] * inv;
    float sn, cs;
    __sincosf(ang, &sn, &cs);
    size_t ib = (size_t)s * DMODEL + h * HDIM + 2 * j;
    float q1 = Q32[ib], q2 = Q32[ib + 1];
    float k1 = K32[ib], k2 = K32[ib + 1];
    size_t ob = ((size_t)h * SEQ + s) * HDIM + 2 * j;
    Q16[ob]     = (_Float16)(q1 * cs - q2 * sn);
    Q16[ob + 1] = (_Float16)(q1 * sn + q2 * cs);
    K16[ob]     = (_Float16)(k1 * cs - k2 * sn);
    K16[ob + 1] = (_Float16)(k1 * sn + k2 * cs);
  }
}

// ---------------------------------------------------------------------------
// Causal flash attention. One wave per (head, 16-query block).
// Block = 256 threads = 8 waves. K-tiles streamed into LDS with
// GLOBAL_LOAD_ASYNC_TO_LDS_B128, next tile overlapped with softmax+PV.
// ---------------------------------------------------------------------------
__global__ __launch_bounds__(256) void attention(
    const _Float16* __restrict__ Q16, const _Float16* __restrict__ K16,
    const _Float16* __restrict__ Vt16, float* __restrict__ attn) {
  __shared__ __align__(16) _Float16 ldsK[8][32 * HDIM];  // 4KB per wave
  __shared__ __align__(16) _Float16 ldsP[8][16 * 32];    // 1KB per wave

  const int tid  = threadIdx.x;
  const int wv   = tid >> 5;
  const int lane = tid & 31;
  const int hi   = lane >> 4;
  const int ko   = hi * 8;
  const int ln   = lane & 15;

  const int Wid = blockIdx.x * 8 + wv;       // 0 .. NHEAD*(SEQ/16)-1
  const int h   = Wid & (NHEAD - 1);
  const int q0  = (Wid >> 4) * 16;
  const int tEnd = q0 + 16;

  const _Float16* Qh = Q16  + ((size_t)h * SEQ + q0) * HDIM;
  const _Float16* Kh = K16  + (size_t)h * SEQ * HDIM;
  const _Float16* Vh = Vt16 + (size_t)h * HDIM * SEQ;
  _Float16* Kbuf = ldsK[wv];
  _Float16* Pbuf = ldsP[wv];

  // Q fragments live in registers for the whole row-block (Dh=64 -> 2 k-steps)
  v16h aQ[2];
#pragma unroll
  for (int kk = 0; kk < 2; ++kk)
    aQ[kk] = load_frag16(Qh + (size_t)ln * HDIM + kk * 32 + ko);

  v8f acc[4] = {};
  float mrow[8], lrow[8];
#pragma unroll
  for (int r = 0; r < 8; ++r) { mrow[r] = -1e30f; lrow[r] = 0.0f; }

  const float scale = 0.125f;   // 1/sqrt(64)

  // Preload first K tile (32 keys x 64 = one contiguous 4KB block of K16)
  async_copy_4k(Kh, Kbuf, lane);

  for (int t0 = 0; t0 < tEnd; t0 += 32) {
    // Wait for the K tile DMA'd into LDS
    asm volatile("s_wait_asynccnt 0x0" ::: "memory");

    // ---- scores: S(16x32) = Q(16x64) @ K^T, 4 WMMAs (K frags from LDS) ----
    v8f sc[2] = {};
    {
      v16h bk[2][2];
#pragma unroll
      for (int j = 0; j < 2; ++j)
#pragma unroll
        for (int kk = 0; kk < 2; ++kk)
          bk[j][kk] = load_frag16(Kbuf + (size_t)(j * 16 + ln) * HDIM
                                       + kk * 32 + ko);
#pragma unroll
      for (int j = 0; j < 2; ++j)
#pragma unroll
        for (int kk = 0; kk < 2; ++kk)
          sc[j] = wmma_f16(aQ[kk], bk[j][kk], sc[j]);
    }

    // Issue async fetch of the NEXT K tile; overlap with softmax + PV below.
    // Fence ds reads of the current tile first so the async engine cannot
    // overwrite LDS under an in-flight ds_load.
    if (t0 + 32 < tEnd) {
      asm volatile("s_wait_dscnt 0x0" ::: "memory");
      async_copy_4k(Kh + (size_t)(t0 + 32) * HDIM, Kbuf, lane);
    }

    // ---- scale + causal mask (arithmetic, no EXEC divergence) ----
    const bool diag = (t0 + 31) > q0;
#pragma unroll
    for (int j = 0; j < 2; ++j)
#pragma unroll
      for (int r = 0; r < 8; ++r) {
        float v = sc[j][r] * scale;
        if (diag) {
          int q = q0 + r + hi * 8;
          int t = t0 + j * 16 + ln;
          if (t > q) v = -1e30f;
        }
        sc[j][r] = v;
      }

    // ---- streaming softmax update (16-lane butterflies, wave32) ----
#pragma unroll
    for (int r = 0; r < 8; ++r) {
      float bm = fmaxf(sc[0][r], sc[1][r]);
#pragma unroll
      for (int msk = 8; msk >= 1; msk >>= 1)
        bm = fmaxf(bm, __shfl_xor(bm, msk, 32));
      float mnew = fmaxf(mrow[r], bm);
      float corr = __expf(mrow[r] - mnew);
      mrow[r] = mnew;
      float p0 = __expf(sc[0][r] - mnew);
      float p1 = __expf(sc[1][r] - mnew);
      sc[0][r] = p0; sc[1][r] = p1;
      float rs = p0 + p1;
#pragma unroll
      for (int msk = 8; msk >= 1; msk >>= 1)
        rs += __shfl_xor(rs, msk, 32);
      lrow[r] = lrow[r] * corr + rs;
#pragma unroll
      for (int j2 = 0; j2 < 4; ++j2) acc[j2][r] *= corr;
    }

    // ---- C-layout -> A-layout via per-wave LDS tile ----
#pragma unroll
    for (int j = 0; j < 2; ++j)
#pragma unroll
      for (int r = 0; r < 8; ++r)
        Pbuf[(r + hi * 8) * 32 + j * 16 + ln] = (_Float16)sc[j][r];
    asm volatile("s_wait_dscnt 0x0" ::: "memory");

    // ---- O += P(16x32) @ V(32x64); V rows contiguous over keys ----
    v16h aP = load_frag16(Pbuf + (size_t)ln * 32 + ko);
#pragma unroll
    for (int j2 = 0; j2 < 4; ++j2) {
      v16h b = load_frag16(Vh + (size_t)(j2 * 16 + ln) * SEQ + t0 + ko);
      acc[j2] = wmma_f16(aP, b, acc[j2]);
    }
  }

  // ---- normalize and scatter into attn [S][DMODEL] ----
#pragma unroll
  for (int j2 = 0; j2 < 4; ++j2)
#pragma unroll
    for (int r = 0; r < 8; ++r) {
      int q = q0 + r + hi * 8;
      int dcol = h * HDIM + j2 * 16 + ln;
      attn[(size_t)q * DMODEL + dcol] = acc[j2][r] / lrow[r];
    }
}

// ---------------------------------------------------------------------------
extern "C" void kernel_launch(void* const* d_in, const int* in_sizes, int n_in,
                              void* d_out, int out_size, void* d_ws, size_t ws_size,
                              hipStream_t stream) {
  (void)in_sizes; (void)n_in; (void)out_size; (void)ws_size;
  const float* x  = (const float*)d_in[0];
  const float* Wq = (const float*)d_in[1];
  const float* Wk = (const float*)d_in[2];
  const float* Wv = (const float*)d_in[3];
  const float* Wo = (const float*)d_in[4];
  const int*  pos = (const int*)d_in[5];
  float* out = (float*)d_out;

  char* ws = (char*)d_ws;
  const size_t MB = 1024ull * 1024ull;
  float*    Q32  = (float*)(ws + 0 * MB);     // 16 MB
  float*    K32  = (float*)(ws + 16 * MB);    // 16 MB
  float*    V32  = (float*)(ws + 32 * MB);    // 16 MB
  _Float16* Q16  = (_Float16*)(ws + 48 * MB); //  8 MB
  _Float16* K16  = (_Float16*)(ws + 56 * MB); //  8 MB
  _Float16* Vt16 = (_Float16*)(ws + 64 * MB); //  8 MB
  float*    attn = Q32;                        // reuse (Q32 dead after rope)

  dim3 gg(SEQ / 128, DMODEL / 64);
  gemm_xwT<<<gg, 256, 0, stream>>>(x, Wq, Q32);
  gemm_xwT<<<gg, 256, 0, stream>>>(x, Wk, K32);
  gemm_xwT<<<gg, 256, 0, stream>>>(x, Wv, V32);

  rope_convert<<<(SEQ * DMODEL) / 256, 256, 0, stream>>>(Q32, K32, V32, pos,
                                                         Q16, K16, Vt16);

  attention<<<(NHEAD * (SEQ / 16)) / 8, 256, 0, stream>>>(Q16, K16, Vt16, attn);

  gemm_xwT<<<gg, 256, 0, stream>>>(attn, Wo, out);
}
